// MMAttn_2250562863585
// MI455X (gfx1250) — matrix operand order repaired and checked by reading, not compiled, not run on hardware
//
#include <hip/hip_runtime.h>
#include <hip/hip_bf16.h>

// ---------------------------------------------------------------------------
// MI455X (gfx1250): wave32, WMMA 16x16x32 f16->f32.
// All GEMM/attention operands pre-converted + pre-swizzled into WMMA fragment
// tile layout (4KB tiles), so hot loops stage tiles with CDNA5 async
// global->LDS b128 copies (ASYNCcnt), double-buffered against WMMA compute.
// ---------------------------------------------------------------------------

typedef __attribute__((ext_vector_type(16))) _Float16 v16h;
typedef __attribute__((ext_vector_type(8)))  _Float16 v8h;
typedef __attribute__((ext_vector_type(4)))  _Float16 v4h;
typedef __attribute__((ext_vector_type(8)))  float    v8f;

#define WMMA16(A, B, C) \
  __builtin_amdgcn_wmma_f32_16x16x32_f16(false, (A), false, (B), (short)0, (C), false, false)

// Async copy 16B global -> LDS (per lane). LDS offset = low 32 bits of the
// generic shared pointer (ISA 10.2: LDS aperture addr[31:0] = LDS offset).
__device__ __forceinline__ void async_ld_b128(void* lds, const void* g) {
  asm volatile("global_load_async_to_lds_b128 %0, %1, off"
               :
               : "v"((unsigned)(size_t)lds), "v"(g)
               : "memory");
}
__device__ __forceinline__ void wait_async0() {
  asm volatile("s_wait_asynccnt 0x0" ::: "memory");
}

// Load one v16h fragment (32B contiguous): two b128 loads.
__device__ __forceinline__ v16h ld_frag(const _Float16* p) {
  v8h lo = *(const v8h*)p;
  v8h hi = *(const v8h*)(p + 8);
  return __builtin_shufflevector(lo, hi, 0, 1, 2, 3, 4, 5, 6, 7,
                                 8, 9, 10, 11, 12, 13, 14, 15);
}

__device__ __forceinline__ float wave_sum32(float v) {
#pragma unroll
  for (int m = 16; m >= 1; m >>= 1) v += __shfl_xor(v, m, 32);
  return v;
}

// Fragment layout maps (16-bit elements, ISA 7.12.2):
//   A (16x32): lane = m + 16*((k&15)>=8), half = 8*(k>=16) + (k&7)
//   B (32x16): lane = n + 16*((k&15)>=8), half = 8*(k>=16) + (k&7)

// ---------------------------------------------------------------------------
// Swizzle A[M,K] f32 row-major -> f16 A-frag tiles:
//   Af[(by*(K/32)+kc)*2048 + strip*512 + lane*16 + half]
// One float4 per thread; grid = M*K/4/256 blocks.
// ---------------------------------------------------------------------------
__global__ __launch_bounds__(256) void swizzle_a(
    const float* __restrict__ A, _Float16* __restrict__ Af, int M, int K) {
  int idx = blockIdx.x * 256 + threadIdx.x;
  int perRow = K >> 2;
  int r  = idx / perRow;
  int k4 = (idx % perRow) << 2;
  const float4 v = *reinterpret_cast<const float4*>(&A[(size_t)r * K + k4]);
  v4h tv;
  tv[0] = (_Float16)v.x; tv[1] = (_Float16)v.y;
  tv[2] = (_Float16)v.z; tv[3] = (_Float16)v.w;
  int by = r >> 6, strip = (r >> 4) & 3, m = r & 15;
  int kc = k4 >> 5, kin = k4 & 31;
  int ln = m + (((kin & 15) >= 8) ? 16 : 0);
  int h0 = ((kin >= 16) ? 8 : 0) + (kin & 7);
  size_t dst = ((size_t)(by * (K >> 5) + kc) * 2048) + strip * 512 + ln * 16 + h0;
  *(v4h*)&Af[dst] = tv;
}

// ---------------------------------------------------------------------------
// Swizzle W[K,N] f32 row-major -> f16 B-frag tiles:
//   Wf[(bx*(K/32)+kc)*2048 + ct*512 + lane*16 + half]
// One 4-deep column per thread; grid = (K/4)*N/256 blocks.
// ---------------------------------------------------------------------------
__global__ __launch_bounds__(256) void swizzle_b(
    const float* __restrict__ W, _Float16* __restrict__ Wf, int K, int N) {
  int idx = blockIdx.x * 256 + threadIdx.x;
  int n   = idx % N;
  int k4  = (idx / N) << 2;
  const float* wp = &W[(size_t)k4 * N + n];
  v4h tv;
  tv[0] = (_Float16)wp[0];
  tv[1] = (_Float16)wp[(size_t)N];
  tv[2] = (_Float16)wp[(size_t)2 * N];
  tv[3] = (_Float16)wp[(size_t)3 * N];
  int bx = n >> 6, ct = (n >> 4) & 3, nn = n & 15;
  int ln = nn + (((k4 & 15) >= 8) ? 16 : 0);
  size_t dst = ((size_t)(bx * (K >> 5) + (k4 >> 5)) * 2048) + ct * 512 + ln * 16 + ((((k4 & 31) >= 16) ? 8 : 0) + (k4 & 7));
  *(v4h*)&Wf[dst] = tv;
}

// ---------------------------------------------------------------------------
// C[M,N] = Af @ Wf + bias, operands pre-swizzled f16 frag tiles.
// Tile 64x64, K-chunk 32, 8 waves/block, wave -> 16x32 C strip.
// Double-buffered async global->LDS staging (1 b128 async per thread/operand).
// ---------------------------------------------------------------------------
__global__ __launch_bounds__(256) void gemm_bias_wmma(
    const _Float16* __restrict__ Af, const _Float16* __restrict__ Wf,
    const float* __restrict__ bias, float* __restrict__ C,
    int M, int N, int K) {
  __shared__ _Float16 AsF[2][2048];
  __shared__ _Float16 BsF[2][2048];

  const int t    = threadIdx.x;
  const int lane = t & 31;
  const int wave = t >> 5;
  const int bx   = blockIdx.x, by = blockIdx.y;
  const int wm   = wave & 3, wn = wave >> 2, m = lane & 15;
  const int KC   = K >> 5;
  const size_t abase = (size_t)by * KC * 2048;
  const size_t wbase = (size_t)bx * KC * 2048;

  v8f c0 = {}, c1 = {};

  async_ld_b128(&AsF[0][t * 8], Af + abase + t * 8);
  async_ld_b128(&BsF[0][t * 8], Wf + wbase + t * 8);
  wait_async0();
  __syncthreads();

  for (int kc = 0; kc < KC; ++kc) {
    const int cur = kc & 1, nxt = cur ^ 1;
    if (kc + 1 < KC) {
      async_ld_b128(&AsF[nxt][t * 8], Af + abase + (size_t)(kc + 1) * 2048 + t * 8);
      async_ld_b128(&BsF[nxt][t * 8], Wf + wbase + (size_t)(kc + 1) * 2048 + t * 8);
      __builtin_prefetch(Af + abase + (size_t)(kc + 2) * 2048 + t * 8, 0, 1);
    }
    v16h a  = ld_frag(&AsF[cur][wm * 512 + lane * 16]);
    v16h b0 = ld_frag(&BsF[cur][(wn * 2 + 0) * 512 + lane * 16]);
    v16h b1 = ld_frag(&BsF[cur][(wn * 2 + 1) * 512 + lane * 16]);
    c0 = WMMA16(a, b0, c0);
    c1 = WMMA16(a, b1, c1);
    wait_async0();
    __syncthreads();
  }

  const int rbase = (lane >= 16) ? 8 : 0;
#pragma unroll
  for (int r = 0; r < 8; ++r) {
    int row  = by * 64 + wm * 16 + r + rbase;
    int col0 = bx * 64 + wn * 32 + m;
    int col1 = col0 + 16;
    C[(size_t)row * N + col0] = c0[r] + bias[col0];
    C[(size_t)row * N + col1] = c1[r] + bias[col1];
  }
}

// ---------------------------------------------------------------------------
// LayerNorm + RoPE + pack f16 directly into fragment-tile layouts:
//   Qs: per (bh, qtile<130>): [dc<2>][lane<32>][half<16>]  (A-frag, 2KB/tile)
//   Ks: per (bh, jtile<65>):  [frag<4>=kh*2+dc][lane][half] (B-frag, 4KB/tile)
//   Vs: per (bh, jtile<65>):  [frag<4>=dsub][lane][half]    (B-frag, 4KB/tile)
// One wave per (b, s, h); lane d handles dims d and d+32 (RoPE pair).
// ---------------------------------------------------------------------------
__global__ __launch_bounds__(256) void ln_rope_pack(
    const float* __restrict__ qkv0, const float* __restrict__ qkv1,
    _Float16* __restrict__ Qs, _Float16* __restrict__ Ks,
    _Float16* __restrict__ Vs) {
  const int lane = threadIdx.x & 31;
  const int unit = blockIdx.x * 8 + (threadIdx.x >> 5);  // < 2*2080*16
  const int h  = unit & 15;
  const int bs = unit >> 4;
  const int s  = bs % 2080;
  const int b  = bs / 2080;
  const int f  = s / 65, j = s % 65;
  const int bh = b * 16 + h;

  const float* row = (j < 64)
      ? (qkv0 + (size_t)(b * 2048 + f * 64 + j) * 3072)
      : (qkv1 + (size_t)(b * 32 + f) * 3072);

  const float inv_freq = __powf(10000.0f, -((float)lane) / 32.0f);
  const float ang = (float)s * inv_freq;
  const float cs = __cosf(ang), sn = __sinf(ang);

  // ---- q: LN + RoPE -> A-frag tiles ----
  {
    float x1 = row[h * 64 + lane], x2 = row[h * 64 + lane + 32];
    float mean = wave_sum32(x1 + x2) * (1.0f / 64.0f);
    float var  = wave_sum32(x1 * x1 + x2 * x2) * (1.0f / 64.0f) - mean * mean;
    float rstd = rsqrtf(var + 1e-5f);
    float n1 = (x1 - mean) * rstd, n2 = (x2 - mean) * rstd;
    float o1 = n1 * cs - n2 * sn, o2 = n1 * sn + n2 * cs;
    int qt = s >> 4, mm = s & 15, k = lane;  // k = d&31 for both halves
    size_t tb  = ((size_t)(bh * 130 + qt) * 2) * 512;
    size_t pos = (size_t)(mm + (((k & 15) >= 8) ? 16 : 0)) * 16 +
                 (((k >= 16) ? 8 : 0) + (k & 7));
    Qs[tb + pos]       = (_Float16)o1;   // dc = 0
    Qs[tb + 512 + pos] = (_Float16)o2;   // dc = 1
  }
  const int jt = s >> 5, key = s & 31;
  const size_t jb = ((size_t)(bh * 65 + jt) * 4) * 512;
  // ---- k: LN + RoPE -> B-frag tiles (contraction dim = d) ----
  {
    float x1 = row[1024 + h * 64 + lane], x2 = row[1024 + h * 64 + lane + 32];
    float mean = wave_sum32(x1 + x2) * (1.0f / 64.0f);
    float var  = wave_sum32(x1 * x1 + x2 * x2) * (1.0f / 64.0f) - mean * mean;
    float rstd = rsqrtf(var + 1e-5f);
    float n1 = (x1 - mean) * rstd, n2 = (x2 - mean) * rstd;
    float o1 = n1 * cs - n2 * sn, o2 = n1 * sn + n2 * cs;
    int kh = key >> 4, n = key & 15, kk = lane;  // kk = d&31
    size_t pos = (size_t)(n + (((kk & 15) >= 8) ? 16 : 0)) * 16 + (((kk >= 16) ? 8 : 0) + (kk & 7));
    Ks[jb + (size_t)(kh * 2 + 0) * 512 + pos] = (_Float16)o1;  // dc = 0
    Ks[jb + (size_t)(kh * 2 + 1) * 512 + pos] = (_Float16)o2;  // dc = 1
  }
  // ---- v -> B-frag tiles (contraction dim = key) ----
  {
    float x1 = row[2048 + h * 64 + lane], x2 = row[2048 + h * 64 + lane + 32];
    int kk = key;
    int d1 = lane, d2 = lane + 32;
    size_t p1 = (size_t)((d1 & 15) + (((kk & 15) >= 8) ? 16 : 0)) * 16 + (((kk >= 16) ? 8 : 0) + (kk & 7));
    size_t p2 = (size_t)((d2 & 15) + (((kk & 15) >= 8) ? 16 : 0)) * 16 + (((kk >= 16) ? 8 : 0) + (kk & 7));
    Vs[jb + (size_t)(d1 >> 4) * 512 + p1] = (_Float16)x1;
    Vs[jb + (size_t)(d2 >> 4) * 512 + p2] = (_Float16)x2;
  }
}

// ---------------------------------------------------------------------------
// Flash attention. Block = 8 waves on same (b,h), wave -> Q tile qb*8+w.
// Per 32-key step: K and V tiles staged by double-buffered async b128 copies
// (contiguous 4KB each, pre-swizzled). 8 WMMA per wave per step.
// ---------------------------------------------------------------------------
__global__ __launch_bounds__(256) void flash_attn_wmma(
    const _Float16* __restrict__ Qs, const _Float16* __restrict__ Ks,
    const _Float16* __restrict__ Vs,
    float* __restrict__ attn0, float* __restrict__ attn1) {
  __shared__ _Float16 KsL[2][2048];
  __shared__ _Float16 VsL[2][2048];
  __shared__ _Float16 PsF[8][512];

  const int t    = threadIdx.x;
  const int lane = t & 31;
  const int wv   = t >> 5;
  const int bh   = blockIdx.x / 17;   // 0..31
  const int qb   = blockIdx.x % 17;   // 0..16
  int qt = qb * 8 + wv; if (qt > 129) qt = 129;
  const int b = bh >> 4, h = bh & 15;
  const int m = lane & 15;

  // ---- Q A-fragments: contiguous, 2x b128 global loads each ----
  const _Float16* qp = Qs + ((size_t)(bh * 130 + qt) * 2) * 512 + lane * 16;
  v16h qa0 = ld_frag(qp);
  v16h qa1 = ld_frag(qp + 512);

  float mrow[8], lrow[8];
  v8f o0 = {}, o1 = {}, o2 = {}, o3 = {};
#pragma unroll
  for (int r = 0; r < 8; ++r) { mrow[r] = -3.0e38f; lrow[r] = 0.0f; }

  const size_t kvbase = (size_t)(bh * 65) * 2048;

  async_ld_b128(&KsL[0][t * 8], Ks + kvbase + t * 8);
  async_ld_b128(&VsL[0][t * 8], Vs + kvbase + t * 8);
  wait_async0();
  __syncthreads();

  for (int jt = 0; jt < 65; ++jt) {
    const int cur = jt & 1, nxt = cur ^ 1;
    if (jt + 1 < 65) {
      async_ld_b128(&KsL[nxt][t * 8], Ks + kvbase + (size_t)(jt + 1) * 2048 + t * 8);
      async_ld_b128(&VsL[nxt][t * 8], Vs + kvbase + (size_t)(jt + 1) * 2048 + t * 8);
    }

    // ---- Q @ K^T : 4 WMMA ----
    v8f s0 = {}, s1 = {};
    {
      v16h kb00 = ld_frag(&KsL[cur][0 * 512 + lane * 16]);
      v16h kb10 = ld_frag(&KsL[cur][1 * 512 + lane * 16]);
      v16h kb01 = ld_frag(&KsL[cur][2 * 512 + lane * 16]);
      v16h kb11 = ld_frag(&KsL[cur][3 * 512 + lane * 16]);
      s0 = WMMA16(qa0, kb00, s0);
      s0 = WMMA16(qa1, kb10, s0);
      s1 = WMMA16(qa0, kb01, s1);
      s1 = WMMA16(qa1, kb11, s1);
    }

    // ---- online softmax ----
    float corr[8];
#pragma unroll
    for (int r = 0; r < 8; ++r) {
      float a = s0[r] * 0.125f, c = s1[r] * 0.125f;
      float tmax = fmaxf(a, c);
#pragma unroll
      for (int mk = 8; mk >= 1; mk >>= 1) tmax = fmaxf(tmax, __shfl_xor(tmax, mk, 32));
      float mn = fmaxf(mrow[r], tmax);
      corr[r] = __expf(mrow[r] - mn);
      mrow[r] = mn;
      float p0 = __expf(a - mn), p1 = __expf(c - mn);
      s0[r] = p0; s1[r] = p1;
      float ps = p0 + p1;
#pragma unroll
      for (int mk = 8; mk >= 1; mk >>= 1) ps += __shfl_xor(ps, mk, 32);
      lrow[r] = lrow[r] * corr[r] + ps;
      o0[r] *= corr[r]; o1[r] *= corr[r]; o2[r] *= corr[r]; o3[r] *= corr[r];
    }

    // ---- P -> A-frag layout in LDS; read back as 2x ds_load_b128 ----
    {
      const int rb   = (lane >= 16) ? 8 : 0;
      const int ph   = m & 7;
      const int pl16 = (m >= 8) ? 16 : 0;
#pragma unroll
      for (int r = 0; r < 8; ++r) {
        int pln = (r + rb) + pl16;
        PsF[wv][pln * 16 + ph]     = (_Float16)s0[r];
        PsF[wv][pln * 16 + 8 + ph] = (_Float16)s1[r];
      }
    }
    v16h pa = ld_frag(&PsF[wv][lane * 16]);

    // ---- P @ V : 4 WMMA ----
    {
      v16h vb0 = ld_frag(&VsL[cur][0 * 512 + lane * 16]);
      v16h vb1 = ld_frag(&VsL[cur][1 * 512 + lane * 16]);
      v16h vb2 = ld_frag(&VsL[cur][2 * 512 + lane * 16]);
      v16h vb3 = ld_frag(&VsL[cur][3 * 512 + lane * 16]);
      o0 = WMMA16(pa, vb0, o0);
      o1 = WMMA16(pa, vb1, o1);
      o2 = WMMA16(pa, vb2, o2);
      o3 = WMMA16(pa, vb3, o3);
    }
    wait_async0();
    __syncthreads();
  }

  // ---- epilogue: normalize, scatter into attn0 / attn1 ----
  const int rb = (lane >= 16) ? 8 : 0;
#pragma unroll
  for (int r = 0; r < 8; ++r) {
    int srow = qt * 16 + r + rb;
    int f = srow / 65, j = srow % 65;
    float invl = 1.0f / lrow[r];
    float* dst;
    size_t row;
    if (j < 64) { dst = attn0; row = (size_t)(b * 2048 + f * 64 + j); }
    else        { dst = attn1; row = (size_t)(b * 32 + f); }
    size_t cb = row * 1024 + (size_t)h * 64 + m;
    dst[cb]      = o0[r] * invl;
    dst[cb + 16] = o1[r] * invl;
    dst[cb + 32] = o2[r] * invl;
    dst[cb + 48] = o3[r] * invl;
  }
}

// ---------------------------------------------------------------------------
extern "C" void kernel_launch(void* const* d_in, const int* in_sizes, int n_in,
                              void* d_out, int out_size, void* d_ws, size_t ws_size,
                              hipStream_t stream) {
  const float* x0    = (const float*)d_in[0];
  const float* x1    = (const float*)d_in[1];
  const float* Wqkv0 = (const float*)d_in[2];
  const float* bqkv0 = (const float*)d_in[3];
  const float* Wqkv1 = (const float*)d_in[4];
  const float* bqkv1 = (const float*)d_in[5];
  const float* Wo0   = (const float*)d_in[6];
  const float* bo0   = (const float*)d_in[7];
  const float* Wo1   = (const float*)d_in[8];
  const float* bo1   = (const float*)d_in[9];
  float* out = (float*)d_out;

  char* ws = (char*)d_ws;
  _Float16* x0f    = (_Float16*)ws;  ws += (size_t)4096 * 1024 * 2;
  _Float16* x1f    = (_Float16*)ws;  ws += (size_t)64 * 1024 * 2;
  _Float16* Wqkv0f = (_Float16*)ws;  ws += (size_t)1024 * 3072 * 2;
  _Float16* Wqkv1f = (_Float16*)ws;  ws += (size_t)1024 * 3072 * 2;
  _Float16* Wo0f   = (_Float16*)ws;  ws += (size_t)1024 * 1024 * 2;
  _Float16* Wo1f   = (_Float16*)ws;  ws += (size_t)1024 * 1024 * 2;
  float* qkv0      = (float*)ws;     ws += (size_t)4096 * 3072 * 4;
  float* qkv1      = (float*)ws;     ws += (size_t)64 * 3072 * 4;
  _Float16* Qs     = (_Float16*)ws;  ws += (size_t)2 * 16 * 2080 * 64 * 2;
  _Float16* Ks     = (_Float16*)ws;  ws += (size_t)2 * 16 * 2080 * 64 * 2;
  _Float16* Vs     = (_Float16*)ws;  ws += (size_t)2 * 16 * 2080 * 64 * 2;
  float* attn0     = (float*)ws;     ws += (size_t)4096 * 1024 * 4;
  float* attn1     = (float*)ws;     ws += (size_t)64 * 1024 * 4;
  _Float16* attn0f = (_Float16*)ws;  ws += (size_t)4096 * 1024 * 2;
  _Float16* attn1f = (_Float16*)ws;  ws += (size_t)64 * 1024 * 2;

  // 0) Pre-convert + pre-swizzle operands to f16 fragment tiles
  swizzle_b<<<3072, 256, 0, stream>>>(Wqkv0, Wqkv0f, 1024, 3072);
  swizzle_b<<<3072, 256, 0, stream>>>(Wqkv1, Wqkv1f, 1024, 3072);
  swizzle_b<<<1024, 256, 0, stream>>>(Wo0, Wo0f, 1024, 1024);
  swizzle_b<<<1024, 256, 0, stream>>>(Wo1, Wo1f, 1024, 1024);
  swizzle_a<<<4096, 256, 0, stream>>>(x0, x0f, 4096, 1024);
  swizzle_a<<<64,   256, 0, stream>>>(x1, x1f, 64, 1024);

  // 1) QKV projections
  gemm_bias_wmma<<<dim3(48, 64), 256, 0, stream>>>(x0f, Wqkv0f, bqkv0, qkv0, 4096, 3072, 1024);
  gemm_bias_wmma<<<dim3(48, 1),  256, 0, stream>>>(x1f, Wqkv1f, bqkv1, qkv1,   64, 3072, 1024);

  // 2) LayerNorm + RoPE + f16 fragment-tile pack
  ln_rope_pack<<<8320, 256, 0, stream>>>(qkv0, qkv1, Qs, Ks, Vs);

  // 3) Flash attention (32 bh x 17 q-tile groups)
  flash_attn_wmma<<<544, 256, 0, stream>>>(Qs, Ks, Vs, attn0, attn1);

  // 4) Output projections -> d_out (y0 then y1)
  swizzle_a<<<4096, 256, 0, stream>>>(attn0, attn0f, 4096, 1024);
  swizzle_a<<<64,   256, 0, stream>>>(attn1, attn1f, 64, 1024);
  gemm_bias_wmma<<<dim3(16, 64), 256, 0, stream>>>(attn0f, Wo0f, bo0, out, 4096, 1024, 1024);
  gemm_bias_wmma<<<dim3(16, 1),  256, 0, stream>>>(attn1f, Wo1f, bo1, out + (size_t)4096 * 1024, 64, 1024, 1024);
}
